// AttentionBlock_51634096832974
// MI455X (gfx1250) — matrix operand order repaired and checked by reading, not compiled, not run on hardware
//
#include <hip/hip_runtime.h>
#include <hip/hip_bf16.h>
#include <math.h>

// ---------------------------------------------------------------------------
// AttentionBlock for MI455X (gfx1250): GN+FiLM folded into QKV GEMM,
// bf16 WMMA for all matmuls, async-to-LDS staging, flash-softmax attention.
// Shapes: B=16, C=512, T=1024, E=1024, heads=8, ch=64, groups=32.
// ---------------------------------------------------------------------------

typedef __bf16 bf16_t;
typedef __attribute__((ext_vector_type(16))) __bf16 v16bf;
typedef __attribute__((ext_vector_type(8)))  __bf16 v8bf;
typedef __attribute__((ext_vector_type(4)))  __bf16 v4bf;
typedef __attribute__((ext_vector_type(8)))  float  v8f;

union Frag16 { v16bf v; v8bf h[2]; };

static __device__ __forceinline__ v8f wmma_bf16(const Frag16& a, const Frag16& b, v8f c) {
  return __builtin_amdgcn_wmma_f32_16x16x32_bf16(false, a.v, false, b.v, (short)0, c,
                                                 false, false);
}

// gfx1250 async copy: 16B global -> LDS, tracked by ASYNCcnt (no VGPR data path)
static __device__ __forceinline__ void async_copy16(void* lds, const void* gptr) {
  unsigned l = (unsigned)(uintptr_t)lds;   // low 32 bits of generic shared addr = LDS offset
  asm volatile("global_load_async_to_lds_b128 %0, %1, off" :: "v"(l), "v"(gptr) : "memory");
}
static __device__ __forceinline__ void async_wait0() {
  asm volatile("s_wait_asynccnt 0x0" ::: "memory");
}

#define NB   16
#define NC   512
#define NT   1024
#define NE   1024
#define NHD  8
#define CH   64
#define TC3  1536

// ---------------------------------------------------------------------------
// K1: FiLM GEMM  film[b, j] = emb_b[j] + silu(emb[b,:]) . emb_w[j,:]
// ---------------------------------------------------------------------------
__global__ __launch_bounds__(256) void film_kernel(
    const float* __restrict__ emb, const float* __restrict__ emb_w,
    const float* __restrict__ emb_b, float* __restrict__ film) {
  __shared__ float se[NE];
  const int b = blockIdx.x, tid = threadIdx.x;
#pragma unroll
  for (int i = 0; i < 4; ++i) {
    int k = tid + 256 * i;
    float v = emb[b * NE + k];
    se[k] = v / (1.0f + __expf(-v));
  }
  __syncthreads();
#pragma unroll
  for (int i = 0; i < 4; ++i) {
    int j = tid + 256 * i;
    float acc = emb_b[j];
    const float4* wr = (const float4*)&emb_w[(size_t)j * NE];
    const float4* sr = (const float4*)se;
    for (int kk = 0; kk < NE / 4; ++kk) {
      float4 w4 = wr[kk], s4 = sr[kk];
      acc += w4.x * s4.x + w4.y * s4.y + w4.z * s4.z + w4.w * s4.w;
    }
    film[b * (2 * NC) + j] = acc;
  }
}

// ---------------------------------------------------------------------------
// K2: GroupNorm statistics. One block per (b, group): 16 ch x 1024 elements.
// ---------------------------------------------------------------------------
__global__ __launch_bounds__(256) void gn_stats_kernel(
    const float* __restrict__ x, float* __restrict__ meanv, float* __restrict__ rstdv) {
  __shared__ float r1[256], r2[256];
  const int tid = threadIdx.x;
  const size_t base = (size_t)blockIdx.x * 16384;
  float s1 = 0.f, s2 = 0.f;
#pragma unroll 8
  for (int i = 0; i < 64; ++i) {
    float v = x[base + tid + 256 * i];
    s1 += v; s2 += v * v;
  }
  r1[tid] = s1; r2[tid] = s2;
  __syncthreads();
  for (int off = 128; off > 0; off >>= 1) {
    if (tid < off) { r1[tid] += r1[tid + off]; r2[tid] += r2[tid + off]; }
    __syncthreads();
  }
  if (tid == 0) {
    float mean = r1[0] * (1.0f / 16384.0f);
    float var = r2[0] * (1.0f / 16384.0f) - mean * mean;
    meanv[blockIdx.x] = mean;
    rstdv[blockIdx.x] = rsqrtf(var + 1e-5f);
  }
}

// ---------------------------------------------------------------------------
// K3: per-(b,c) affine coefficients:  h = x*A + B  (GN + FiLM folded)
// ---------------------------------------------------------------------------
__global__ __launch_bounds__(256) void coeff_kernel(
    const float* __restrict__ film, const float* __restrict__ meanv,
    const float* __restrict__ rstdv, const float* __restrict__ gamma,
    const float* __restrict__ beta, float* __restrict__ Agn, float* __restrict__ Bcv) {
  int idx = blockIdx.x * 256 + threadIdx.x;   // 16*512
  int b = idx >> 9, c = idx & 511, g = c >> 4;
  float mean = meanv[b * 32 + g], rs = rstdv[b * 32 + g];
  float sc = film[b * (2 * NC) + c], sh = film[b * (2 * NC) + NC + c];
  float ga = gamma[c], be = beta[c];
  Agn[idx] = rs * ga * (1.0f + sc);
  Bcv[idx] = (be - mean * rs * ga) * (1.0f + sc) + sh;
}

// ---------------------------------------------------------------------------
// K4: folded QKV bias:  bias2[b,o] = qkv_b[o] + sum_c qkv_w[o,c] * B[b,c]
// ---------------------------------------------------------------------------
__global__ __launch_bounds__(256) void qkv_bias_kernel(
    const float* __restrict__ qkv_w, const float* __restrict__ qkv_b,
    const float* __restrict__ Bcv, float* __restrict__ bias2) {
  __shared__ float sb[NC];
  const int b = blockIdx.x, tid = threadIdx.x;
  sb[tid] = Bcv[b * NC + tid];
  sb[tid + 256] = Bcv[b * NC + tid + 256];
  __syncthreads();
#pragma unroll
  for (int i = 0; i < 6; ++i) {
    int o = tid + 256 * i;
    float acc = qkv_b[o];
    const float4* wr = (const float4*)&qkv_w[(size_t)o * NC];
    const float4* sr = (const float4*)sb;
    for (int kk = 0; kk < NC / 4; ++kk) {
      float4 w4 = wr[kk], s4 = sr[kk];
      acc += w4.x * s4.x + w4.y * s4.y + w4.z * s4.z + w4.w * s4.w;
    }
    bias2[b * TC3 + o] = acc;
  }
}

// ---------------------------------------------------------------------------
// K5: transpose+convert x -> xT bf16 [b][t][c]  (LDS-tiled 64x64)
// ---------------------------------------------------------------------------
__global__ __launch_bounds__(256) void xT_kernel(
    const float* __restrict__ x, bf16_t* __restrict__ xT) {
  __shared__ bf16_t tile[64 * 72];
  const int b = blockIdx.z, cbase = blockIdx.y * 64, tbase = blockIdx.x * 64;
  const int tid = threadIdx.x;
#pragma unroll
  for (int i = 0; i < 4; ++i) {               // read along t (coalesced float4)
    int idx = tid + 256 * i;
    int cl = idx >> 4, tv = idx & 15;
    float4 v4 = *(const float4*)&x[((size_t)b * NC + cbase + cl) * NT + tbase + tv * 4];
    v4bf o4 = { (bf16_t)v4.x, (bf16_t)v4.y, (bf16_t)v4.z, (bf16_t)v4.w };
    *(v4bf*)&tile[cl * 72 + tv * 4] = o4;
  }
  __syncthreads();
#pragma unroll
  for (int i = 0; i < 4; ++i) {               // write along c (coalesced 8B)
    int idx = tid + 256 * i;
    int tl = idx >> 4, cv = idx & 15;
    v4bf o4 = { tile[(cv * 4 + 0) * 72 + tl], tile[(cv * 4 + 1) * 72 + tl],
                tile[(cv * 4 + 2) * 72 + tl], tile[(cv * 4 + 3) * 72 + tl] };
    *(v4bf*)&xT[((size_t)b * NT + tbase + tl) * NC + cbase + cv * 4] = o4;
  }
}

// ---------------------------------------------------------------------------
// K6: proj_w fp32 -> bf16
// ---------------------------------------------------------------------------
__global__ __launch_bounds__(256) void projw_cvt_kernel(
    const float* __restrict__ pw, bf16_t* __restrict__ pwb) {
  int idx = blockIdx.x * 256 + threadIdx.x;   // 65536 threads * 4
  float4 v4 = *(const float4*)&pw[(size_t)idx * 4];
  v4bf o4 = { (bf16_t)v4.x, (bf16_t)v4.y, (bf16_t)v4.z, (bf16_t)v4.w };
  *(v4bf*)&pwb[(size_t)idx * 4] = o4;
}

// ---------------------------------------------------------------------------
// K7: QKV GEMM (bf16 WMMA, async-to-LDS B staging).
//   qkv[b,o,t] = sum_c (qkv_w[o,c]*A[b,c]) xT[b,t,c] + bias2[b,o]
//   q/k channels -> qkT[b][t][h*128 + r]   (fragment-native layout)
//   v channels   -> vws[b][h*64 + c][t]
// ---------------------------------------------------------------------------
__global__ __launch_bounds__(256) void qkv_gemm_kernel(
    const bf16_t* __restrict__ xT, const float* __restrict__ qkv_w,
    const float* __restrict__ Agn, const float* __restrict__ bias2,
    bf16_t* __restrict__ qkT, bf16_t* __restrict__ vws) {
  __shared__ bf16_t Alds[128 * 32];      // [m][k]
  __shared__ bf16_t Blds[128 * 40];      // [n][k], 80B rows (16B-aligned)
  const int b = blockIdx.z;
  const int mbase = blockIdx.y * 128, nbase = blockIdx.x * 128;
  const int tid = threadIdx.x, wave = tid >> 5, lane = tid & 31;
  const int wm = wave >> 2, wn = wave & 3;
  const int hl = lane >> 4, lq = lane & 15;

  v8f acc[4][2];
#pragma unroll
  for (int m = 0; m < 4; ++m)
#pragma unroll
    for (int n = 0; n < 2; ++n) acc[m][n] = (v8f){};

  for (int kb = 0; kb < NC; kb += 32) {
    __syncthreads();
    // A: qkv_w row * Agn -> bf16 (float4 reads, 8B LDS stores)
#pragma unroll
    for (int i = 0; i < 4; ++i) {
      int idx = tid + 256 * i;
      int ml = idx >> 3, kv = idx & 7;
      float4 w4 = *(const float4*)&qkv_w[(size_t)(mbase + ml) * NC + kb + kv * 4];
      const float* ag = &Agn[b * NC + kb + kv * 4];
      v4bf o4 = { (bf16_t)(w4.x * ag[0]), (bf16_t)(w4.y * ag[1]),
                  (bf16_t)(w4.z * ag[2]), (bf16_t)(w4.w * ag[3]) };
      *(v4bf*)&Alds[ml * 32 + kv * 4] = o4;
    }
    // B: async bf16 copy from xT[b][t][c] (already fragment-major)
#pragma unroll
    for (int i = 0; i < 2; ++i) {
      int idx = tid + 256 * i;
      int nl = idx >> 2, seg = idx & 3;
      async_copy16(&Blds[nl * 40 + seg * 8],
                   &xT[((size_t)b * NT + nbase + nl) * NC + kb + seg * 8]);
    }
    if (kb + 32 < NC)
      __builtin_prefetch(&qkv_w[(size_t)(mbase + (tid >> 1)) * NC + kb + 32], 0, 0);
    async_wait0();
    __syncthreads();

    Frag16 af[4], bfr[2];
#pragma unroll
    for (int m = 0; m < 4; ++m) {
      int row = wm * 64 + m * 16 + lq;       // A lanes: K {0..7,16..23}/{8..15,24..31}
      af[m].h[0] = *(const v8bf*)&Alds[row * 32 + hl * 8];
      af[m].h[1] = *(const v8bf*)&Alds[row * 32 + 16 + hl * 8];
    }
#pragma unroll
    for (int n = 0; n < 2; ++n) {
      int col = wn * 32 + n * 16 + lq;       // B lanes: K {0..7,16..23}/{8..15,24..31}
      bfr[n].h[0] = *(const v8bf*)&Blds[col * 40 + hl * 8];
      bfr[n].h[1] = *(const v8bf*)&Blds[col * 40 + 16 + hl * 8];
    }
#pragma unroll
    for (int m = 0; m < 4; ++m)
#pragma unroll
      for (int n = 0; n < 2; ++n) acc[m][n] = wmma_bf16(af[m], bfr[n], acc[m][n]);
  }

#pragma unroll
  for (int m = 0; m < 4; ++m) {
    int o0 = mbase + wm * 64 + m * 16;       // 16-row subtile lies in one q/k/v segment
    int hh = o0 / 192, r0 = o0 - hh * 192;
    if (r0 < 128) {                          // q or k -> qkT[b][t][hh*128 + r]
#pragma unroll
      for (int n = 0; n < 2; ++n)
#pragma unroll
        for (int r = 0; r < 8; ++r) {
          int t = nbase + wn * 32 + n * 16 + lq;
          int col = hh * 128 + r0 + r + 8 * hl;
          float v = acc[m][n][r] + bias2[b * TC3 + o0 + r + 8 * hl];
          qkT[((size_t)b * NT + t) * 1024 + col] = (bf16_t)v;
        }
    } else {                                 // v -> vws[b][hh*64 + c][t]
      int vc0 = hh * 64 + (r0 - 128);
#pragma unroll
      for (int n = 0; n < 2; ++n)
#pragma unroll
        for (int r = 0; r < 8; ++r) {
          int t = nbase + wn * 32 + n * 16 + lq;
          float v = acc[m][n][r] + bias2[b * TC3 + o0 + r + 8 * hl];
          vws[((size_t)b * NC + vc0 + r + 8 * hl) * NT + t] = (bf16_t)v;
        }
    }
  }
}

// ---------------------------------------------------------------------------
// K8: flash attention. grid (qtile=8, head=8, b=16), 8 waves x 16 queries.
// Q frags direct from qkT; K chunks async-staged; V frags direct from vws.
// ---------------------------------------------------------------------------
__global__ __launch_bounds__(256) void attn_kernel(
    const bf16_t* __restrict__ qkT, const bf16_t* __restrict__ vws,
    bf16_t* __restrict__ aT) {
  __shared__ bf16_t Klds[64 * 72];           // [ts_local][c], 144B rows
  __shared__ bf16_t Plds[8 * 16 * 72];       // per-wave [q_local][s]
  const int b = blockIdx.z, h = blockIdx.y, qt = blockIdx.x;
  const int tid = threadIdx.x, wave = tid >> 5, lane = tid & 31;
  const int hl = lane >> 4, lq = lane & 15;
  const bf16_t* vp = vws + ((size_t)b * NC + h * CH) * NT;

  // Q A-fragments straight from qkT[b][t][h*128 + c]
  Frag16 qa[2];
  {
    const bf16_t* qrow = qkT + ((size_t)b * NT + qt * 128 + wave * 16 + lq) * 1024 + h * 128;
#pragma unroll
    for (int st = 0; st < 2; ++st) {
      qa[st].h[0] = *(const v8bf*)&qrow[st * 32 + hl * 8];
      qa[st].h[1] = *(const v8bf*)&qrow[st * 32 + 16 + hl * 8];
    }
  }

  float mrow[8], lrow[8];
  v8f oacc[4];
#pragma unroll
  for (int r = 0; r < 8; ++r) { mrow[r] = -1e30f; lrow[r] = 0.f; }
#pragma unroll
  for (int j = 0; j < 4; ++j) oacc[j] = (v8f){};

  for (int ts = 0; ts < NT; ts += 64) {
    __syncthreads();
#pragma unroll
    for (int i = 0; i < 2; ++i) {            // async-stage K chunk [64 keys][64 c]
      int idx = tid + 256 * i;
      int row = idx >> 3, seg = idx & 7;
      async_copy16(&Klds[row * 72 + seg * 8],
                   &qkT[((size_t)b * NT + ts + row) * 1024 + h * 128 + 64 + seg * 8]);
    }
    async_wait0();
    __syncthreads();

    // ---- S = (Q^T K) * 0.125 : 4 N-subtiles of 16 keys
    v8f s[4];
#pragma unroll
    for (int j = 0; j < 4; ++j) {
      Frag16 kb0, kb1;
      int col = j * 16 + lq;
      kb0.h[0] = *(const v8bf*)&Klds[col * 72 + hl * 8];
      kb0.h[1] = *(const v8bf*)&Klds[col * 72 + 16 + hl * 8];
      kb1.h[0] = *(const v8bf*)&Klds[col * 72 + 32 + hl * 8];
      kb1.h[1] = *(const v8bf*)&Klds[col * 72 + 48 + hl * 8];
      v8f z = (v8f){};
      s[j] = wmma_bf16(qa[0], kb0, z);
      s[j] = wmma_bf16(qa[1], kb1, s[j]);
      s[j] = s[j] * 0.125f;                  // (1/64^(1/4))^2 on logits
    }

    // ---- online softmax (row M = r + 8*half lives in one 16-lane half)
#pragma unroll
    for (int r = 0; r < 8; ++r) {
      float tm = fmaxf(fmaxf(s[0][r], s[1][r]), fmaxf(s[2][r], s[3][r]));
#pragma unroll
      for (int off = 8; off >= 1; off >>= 1) tm = fmaxf(tm, __shfl_xor(tm, off, 16));
      float nm = fmaxf(mrow[r], tm);
      float corr = __expf(mrow[r] - nm);
      mrow[r] = nm;
      float rs = 0.f;
#pragma unroll
      for (int j = 0; j < 4; ++j) {
        float p = __expf(s[j][r] - nm);
        s[j][r] = p; rs += p;
      }
#pragma unroll
      for (int off = 8; off >= 1; off >>= 1) rs += __shfl_xor(rs, off, 16);
      lrow[r] = lrow[r] * corr + rs;
#pragma unroll
      for (int jc = 0; jc < 4; ++jc) oacc[jc][r] *= corr;
    }

    // ---- D-layout P -> A-layout via wave-private LDS
    bf16_t* pl = &Plds[wave * 16 * 72];
#pragma unroll
    for (int j = 0; j < 4; ++j)
#pragma unroll
      for (int r = 0; r < 8; ++r)
        pl[(r + 8 * hl) * 72 + j * 16 + lq] = (bf16_t)s[j][r];
    Frag16 pa[2];
#pragma unroll
    for (int st = 0; st < 2; ++st) {
      int sb = st * 32;
      pa[st].h[0] = *(const v8bf*)&pl[lq * 72 + sb + hl * 8];
      pa[st].h[1] = *(const v8bf*)&pl[lq * 72 + sb + 16 + hl * 8];
    }

    // ---- O += P V^T ; V^T B-fragments straight from global (K=s contiguous)
#pragma unroll
    for (int jc = 0; jc < 4; ++jc) {
      Frag16 vb0, vb1;
      const bf16_t* vrow = vp + (size_t)(jc * 16 + lq) * NT + ts;
      vb0.h[0] = *(const v8bf*)&vrow[hl * 8];
      vb0.h[1] = *(const v8bf*)&vrow[16 + hl * 8];
      vb1.h[0] = *(const v8bf*)&vrow[32 + hl * 8];
      vb1.h[1] = *(const v8bf*)&vrow[48 + hl * 8];
      oacc[jc] = wmma_bf16(pa[0], vb0, oacc[jc]);
      oacc[jc] = wmma_bf16(pa[1], vb1, oacc[jc]);
    }
  }

  // finalize: divide by row sum, store aT[b][t][c] bf16 (proj-fragment-major)
#pragma unroll
  for (int jc = 0; jc < 4; ++jc)
#pragma unroll
    for (int r = 0; r < 8; ++r) {
      float v = oacc[jc][r] / lrow[r];
      int c = h * CH + jc * 16 + lq;
      int t = qt * 128 + wave * 16 + r + 8 * hl;
      aT[((size_t)b * NT + t) * NC + c] = (bf16_t)v;
    }
}

// ---------------------------------------------------------------------------
// K9: proj GEMM + residual (A and B both async-staged bf16).
//   out[b,o,t] = x + proj_b[o] + sum_c proj_w[o,c] * aT[b,t,c]
// ---------------------------------------------------------------------------
__global__ __launch_bounds__(256) void proj_kernel(
    const bf16_t* __restrict__ aT, const bf16_t* __restrict__ pwb,
    const float* __restrict__ proj_b, const float* __restrict__ x,
    float* __restrict__ out) {
  __shared__ bf16_t Alds[128 * 40];
  __shared__ bf16_t Blds[128 * 40];
  const int b = blockIdx.z;
  const int mbase = blockIdx.y * 128, nbase = blockIdx.x * 128;
  const int tid = threadIdx.x, wave = tid >> 5, lane = tid & 31;
  const int wm = wave >> 2, wn = wave & 3;
  const int hl = lane >> 4, lq = lane & 15;

  v8f acc[4][2];
#pragma unroll
  for (int m = 0; m < 4; ++m)
#pragma unroll
    for (int n = 0; n < 2; ++n) acc[m][n] = (v8f){};

  for (int kb = 0; kb < NC; kb += 32) {
    __syncthreads();
#pragma unroll
    for (int i = 0; i < 2; ++i) {            // A: proj_w bf16, async
      int idx = tid + 256 * i;
      int ml = idx >> 2, seg = idx & 3;
      async_copy16(&Alds[ml * 40 + seg * 8],
                   &pwb[(size_t)(mbase + ml) * NC + kb + seg * 8]);
    }
#pragma unroll
    for (int i = 0; i < 2; ++i) {            // B: aT, async
      int idx = tid + 256 * i;
      int nl = idx >> 2, seg = idx & 3;
      async_copy16(&Blds[nl * 40 + seg * 8],
                   &aT[((size_t)b * NT + nbase + nl) * NC + kb + seg * 8]);
    }
    async_wait0();
    __syncthreads();

    Frag16 af[4], bfr[2];
#pragma unroll
    for (int m = 0; m < 4; ++m) {
      int row = wm * 64 + m * 16 + lq;
      af[m].h[0] = *(const v8bf*)&Alds[row * 40 + hl * 8];
      af[m].h[1] = *(const v8bf*)&Alds[row * 40 + 16 + hl * 8];
    }
#pragma unroll
    for (int n = 0; n < 2; ++n) {
      int col = wn * 32 + n * 16 + lq;
      bfr[n].h[0] = *(const v8bf*)&Blds[col * 40 + hl * 8];
      bfr[n].h[1] = *(const v8bf*)&Blds[col * 40 + 16 + hl * 8];
    }
#pragma unroll
    for (int m = 0; m < 4; ++m)
#pragma unroll
      for (int n = 0; n < 2; ++n) acc[m][n] = wmma_bf16(af[m], bfr[n], acc[m][n]);
  }

#pragma unroll
  for (int m = 0; m < 4; ++m)
#pragma unroll
    for (int n = 0; n < 2; ++n)
#pragma unroll
      for (int r = 0; r < 8; ++r) {
        int o = mbase + wm * 64 + m * 16 + r + 8 * hl;
        int t = nbase + wn * 32 + n * 16 + lq;
        size_t xi = ((size_t)b * NC + o) * NT + t;
        out[xi] = x[xi] + proj_b[o] + acc[m][n][r];
      }
}

// ---------------------------------------------------------------------------
extern "C" void kernel_launch(void* const* d_in, const int* in_sizes, int n_in,
                              void* d_out, int out_size, void* d_ws, size_t ws_size,
                              hipStream_t stream) {
  const float* x        = (const float*)d_in[0];   // [16,512,32,32]
  const float* emb      = (const float*)d_in[1];   // [16,1024]
  const float* gn_gamma = (const float*)d_in[2];   // [512]
  const float* gn_beta  = (const float*)d_in[3];   // [512]
  const float* emb_w    = (const float*)d_in[4];   // [1024,1024]
  const float* emb_b    = (const float*)d_in[5];   // [1024]
  const float* qkv_w    = (const float*)d_in[6];   // [1536,512]
  const float* qkv_b    = (const float*)d_in[7];   // [1536]
  const float* proj_w   = (const float*)d_in[8];   // [512,512]
  const float* proj_b   = (const float*)d_in[9];   // [512]
  float* out = (float*)d_out;

  char* w = (char*)d_ws;
  float*  film  = (float*)(w + 0);          //    65536 B
  float*  meanv = (float*)(w + 65536);      //     2048 B
  float*  rstdv = (float*)(w + 67584);      //     2048 B
  float*  Agn   = (float*)(w + 69632);      //    32768 B
  float*  Bcv   = (float*)(w + 102400);     //    32768 B
  float*  bias2 = (float*)(w + 135168);     //    98304 B
  bf16_t* pwb   = (bf16_t*)(w + 233472);    //   524288 B
  bf16_t* xT    = (bf16_t*)(w + 757760);    // 16777216 B
  bf16_t* qkT   = (bf16_t*)(w + 17534976);  // 33554432 B
  bf16_t* vws   = (bf16_t*)(w + 51089408);  // 16777216 B
  bf16_t* aT    = (bf16_t*)(w + 67866624);  // 16777216 B  (total ~80.7 MiB)

  film_kernel    <<<NB, 256, 0, stream>>>(emb, emb_w, emb_b, film);
  gn_stats_kernel<<<NB * 32, 256, 0, stream>>>(x, meanv, rstdv);
  xT_kernel      <<<dim3(NT / 64, NC / 64, NB), 256, 0, stream>>>(x, xT);
  projw_cvt_kernel<<<256, 256, 0, stream>>>(proj_w, pwb);
  coeff_kernel   <<<32, 256, 0, stream>>>(film, meanv, rstdv, gn_gamma, gn_beta, Agn, Bcv);
  qkv_bias_kernel<<<NB, 256, 0, stream>>>(qkv_w, qkv_b, Bcv, bias2);
  qkv_gemm_kernel<<<dim3(NT / 128, TC3 / 128, NB), 256, 0, stream>>>(xT, qkv_w, Agn, bias2, qkT, vws);
  attn_kernel    <<<dim3(NT / 128, NHD, NB), 256, 0, stream>>>(qkT, vws, aT);
  proj_kernel    <<<dim3(NT / 128, NC / 128, NB), 256, 0, stream>>>(aT, pwb, proj_b, x, out);
}
